// GptOssDecoderLayer_4922032521856
// MI455X (gfx1250) — compile-verified, matrix operand order checked by reading, not run on hardware
//
#include <hip/hip_runtime.h>
#include <hip/hip_bf16.h>
#include <math.h>

// ---------------- problem constants (match reference) ----------------
#define T_TOK 2048
#define DM    2048
#define NH    32
#define NKV   8
#define HD    64
#define NE    16
#define TOPK  4
#define FF    2048
#define EPS_R 1e-6f
#define THETA 10000.0f

typedef __attribute__((ext_vector_type(16))) _Float16 v16h;
typedef __attribute__((ext_vector_type(8)))  _Float16 v8h;
typedef __attribute__((ext_vector_type(8)))  float    v8f;

// =====================================================================
// WMMA fragment mapping (CDNA5 wave32, v_wmma_f32_16x16x32_f16):
//   A (16x32 MxK): lane holds M=lane&15; element j holds
//       K = (j<8 ? j : j+8) + 8*(lane>=16)   -> two runs of 8 consecutive K
//   B (32x16 KxN): lane holds N=lane&15; same K mapping.
//   C/D f32: VGPR r, lane -> (M = r + 8*(lane>=16), N = lane&15).
// LDS staging: A row-major [16][SA_LD], B col-major [128][SA_LD]; both padded
// to 40 halves (80B) so every fragment is two aligned ds_load_b128.
// =====================================================================
#define SA_LD 40

__device__ __forceinline__ v8f wmma_frag_mma(const _Float16* __restrict__ sA, int aOff,
                                             const _Float16* __restrict__ sBt, int bOff,
                                             v8f acc) {
    const v8h alo = *(const v8h*)(sA + aOff);
    const v8h ahi = *(const v8h*)(sA + aOff + 16);
    const v8h blo = *(const v8h*)(sBt + bOff);
    const v8h bhi = *(const v8h*)(sBt + bOff + 16);
    v16h a, b;
#pragma unroll
    for (int j = 0; j < 8; ++j) {
        a[j] = alo[j]; a[j + 8] = ahi[j];
        b[j] = blo[j]; b[j + 8] = bhi[j];
    }
    return __builtin_amdgcn_wmma_f32_16x16x32_f16(false, a, false, b,
                                                  (short)0, acc, false, false);
}

// store one global B row segment (16 halves) into col-major LDS tile
__device__ __forceinline__ void stage_b_colmajor(_Float16* __restrict__ sBt,
                                                 const _Float16* __restrict__ g,
                                                 int br, int bs) {
    const v8h b0 = *(const v8h*)(g);
    const v8h b1 = *(const v8h*)(g + 8);
#pragma unroll
    for (int i = 0; i < 8; ++i) {
        sBt[(bs * 16 + i)     * SA_LD + br] = b0[i];
        sBt[(bs * 16 + 8 + i) * SA_LD + br] = b1[i];
    }
}

// ---------------- fp32 -> f16 conversion (weights / one-time) ----------------
__global__ void f32_to_f16(const float* __restrict__ src, _Float16* __restrict__ dst,
                           size_t n) {
    size_t i = (size_t)blockIdx.x * blockDim.x + threadIdx.x;
    const size_t stride = (size_t)gridDim.x * blockDim.x;
    for (; i < n; i += stride) dst[i] = (_Float16)src[i];
}

// =====================================================================
// LDS-staged GEMM: C[M,N](f32) = Ah[M,K](f16) @ Bh[K,N](f16) (+res if RES)
// block 256 thr = 8 waves; tile M16 x N128; K-step 32.
// =====================================================================
template <bool RES>
__global__ void gemm_h_wmma(const _Float16* __restrict__ A, const _Float16* __restrict__ B,
                            const float* __restrict__ res, float* __restrict__ C,
                            int N, int K) {
    __shared__ __align__(16) _Float16 sA[16 * SA_LD];
    __shared__ __align__(16) _Float16 sBt[128 * SA_LD];
    const int t = threadIdx.x;
    const int lane = t & 31, wave = t >> 5;
    const int half = lane >> 4, l16 = lane & 15;
    const int m0 = blockIdx.y * 16;
    const int nblk = blockIdx.x * 128;
    const int ar = t >> 4, ap = t & 15;   // A loader: row, half-pair
    const int br = t >> 3, bs = t & 7;    // B loader: row, 16-half segment
    const int aOff = l16 * SA_LD + half * 8;
    const int bOff = (wave * 16 + l16) * SA_LD + half * 8;
    v8f acc = {};
    for (int kc = 0; kc < K; kc += 32) {
        *(unsigned*)(sA + ar * SA_LD + ap * 2) =
            *(const unsigned*)(A + (size_t)(m0 + ar) * K + kc + ap * 2);
        stage_b_colmajor(sBt, B + (size_t)(kc + br) * N + nblk + bs * 16, br, bs);
        __syncthreads();
        acc = wmma_frag_mma(sA, aOff, sBt, bOff, acc);
        __syncthreads();
    }
    const int col = nblk + wave * 16 + l16;
#pragma unroll
    for (int r = 0; r < 8; ++r) {
        const int row = m0 + r + half * 8;
        const size_t idx = (size_t)row * N + col;
        C[idx] = RES ? (acc[r] + res[idx]) : acc[r];
    }
}

// ---------------- RMSNorm (f32 in, optional f32 + f16 out) ----------------
__global__ void rmsnorm_kernel(const float* __restrict__ x, const float* __restrict__ sc,
                               float* __restrict__ yf, _Float16* __restrict__ yh) {
    __shared__ float red[256];
    const int t = blockIdx.x;
    const float* xr = x + (size_t)t * DM;
    float s = 0.f;
    for (int d = threadIdx.x; d < DM; d += 256) { float v = xr[d]; s += v * v; }
    red[threadIdx.x] = s; __syncthreads();
    for (int st = 128; st > 0; st >>= 1) {
        if (threadIdx.x < st) red[threadIdx.x] += red[threadIdx.x + st];
        __syncthreads();
    }
    const float inv = rsqrtf(red[0] / (float)DM + EPS_R);
    for (int d = threadIdx.x; d < DM; d += 256) {
        const float v = xr[d] * inv * sc[d];
        if (yf) yf[(size_t)t * DM + d] = v;
        if (yh) yh[(size_t)t * DM + d] = (_Float16)v;
    }
}

// ---------------- NeoX RoPE in-place on f32 [T, nH, HD] ----------------
__global__ void rope_kernel(float* __restrict__ x, int nH,
                            const int* __restrict__ positions) {
    const int half = HD / 2;
    const int i = blockIdx.x * blockDim.x + threadIdx.x;
    const int total = T_TOK * nH * half;
    if (i >= total) return;
    const int d = i % half;
    const int h = (i / half) % nH;
    const int t = i / (half * nH);
    const float invf = __powf(THETA, -(float)d / (float)half);
    const float ang = (float)positions[t] * invf;
    const float c = __cosf(ang), s = __sinf(ang);
    float* p = x + ((size_t)t * nH + h) * HD;
    const float x1 = p[d], x2 = p[d + half];
    p[d]        = x1 * c - x2 * s;
    p[d + half] = x2 * c + x1 * s;
}

// ---------------- causal GQA attention: one wave per (head, query) ----------------
__global__ void attn_kernel(const float* __restrict__ q, const float* __restrict__ k,
                            const float* __restrict__ v, _Float16* __restrict__ ctxh) {
    const int h = blockIdx.x;
    const int t = blockIdx.y;
    const int lane = threadIdx.x;          // 2 head-dims per lane
    const int kvh = h / (NH / NKV);
    const int d0 = lane * 2;
    const float* qp = q + ((size_t)t * NH + h) * HD;
    const float q0 = qp[d0], q1 = qp[d0 + 1];
    float m = -3.4e38f, l = 0.f, a0 = 0.f, a1 = 0.f;
    for (int kk = 0; kk <= t; ++kk) {
        const float* kp = k + ((size_t)kk * NKV + kvh) * HD;
        float part = q0 * kp[d0] + q1 * kp[d0 + 1];
#pragma unroll
        for (int off = 16; off >= 1; off >>= 1) part += __shfl_xor(part, off, 32);
        const float s = part * 0.125f;     // HD^-0.5
        const float mn = fmaxf(m, s);
        const float corr = __expf(m - mn);
        const float p = __expf(s - mn);
        const float* vp = v + ((size_t)kk * NKV + kvh) * HD;
        l  = l  * corr + p;
        a0 = a0 * corr + p * vp[d0];
        a1 = a1 * corr + p * vp[d0 + 1];
        m = mn;
    }
    _Float16* op = ctxh + ((size_t)t * NH + h) * HD;
    op[d0]     = (_Float16)(a0 / l);
    op[d0 + 1] = (_Float16)(a1 / l);
}

// ---------------- router logits (small; f32) ----------------
__global__ void router_kernel(const float* __restrict__ hn2, const float* __restrict__ rw,
                              const float* __restrict__ rb, float* __restrict__ logits) {
    const int i = blockIdx.x * blockDim.x + threadIdx.x;
    if (i >= T_TOK * NE) return;
    const int t = i / NE, e = i % NE;
    float s = rb[e];
    const float* hr = hn2 + (size_t)t * DM;
    for (int d = 0; d < DM; ++d) s += hr[d] * rw[(size_t)d * NE + e];
    logits[i] = s;
}

__global__ void zero_ints(int* __restrict__ p, int n) {
    const int i = blockIdx.x * blockDim.x + threadIdx.x;
    if (i < n) p[i] = 0;
}

// ---------------- softmax + top-k + renorm + expert counts ----------------
__global__ void topk_kernel(const float* __restrict__ logits, int* __restrict__ ids,
                            float* __restrict__ wts, int* __restrict__ counts) {
    const int t = blockIdx.x * blockDim.x + threadIdx.x;
    if (t >= T_TOK) return;
    float p[NE]; float mx = -3.4e38f;
    for (int e = 0; e < NE; ++e) { p[e] = logits[t * NE + e]; mx = fmaxf(mx, p[e]); }
    float sum = 0.f;
    for (int e = 0; e < NE; ++e) { p[e] = __expf(p[e] - mx); sum += p[e]; }
    for (int e = 0; e < NE; ++e) p[e] /= sum;
    bool used[NE]; for (int e = 0; e < NE; ++e) used[e] = false;
    float wsum = 0.f;
    for (int j = 0; j < TOPK; ++j) {
        int best = 0; float bv = -1.f;
        for (int e = 0; e < NE; ++e)
            if (!used[e] && p[e] > bv) { bv = p[e]; best = e; }
        used[best] = true;
        ids[t * TOPK + j] = best;
        wts[t * TOPK + j] = bv;
        wsum += bv;
    }
    for (int j = 0; j < TOPK; ++j) {
        wts[t * TOPK + j] /= wsum;
        atomicAdd(&counts[ids[t * TOPK + j]], 1);
    }
}

__global__ void offsets_kernel(const int* __restrict__ counts, int* __restrict__ offsets,
                               int* __restrict__ fill) {
    if (threadIdx.x == 0) {
        offsets[0] = 0;
        for (int e = 0; e < NE; ++e) {
            offsets[e + 1] = offsets[e] + counts[e];
            fill[e] = 0;
        }
    }
}

__global__ void scatter_kernel(const int* __restrict__ ids, const float* __restrict__ wts,
                               const int* __restrict__ offsets, int* __restrict__ fill,
                               int* __restrict__ rowToken, int* __restrict__ assignOf,
                               float* __restrict__ wArr) {
    const int t = blockIdx.x * blockDim.x + threadIdx.x;
    if (t >= T_TOK) return;
    for (int j = 0; j < TOPK; ++j) {
        const int e = ids[t * TOPK + j];
        const int idx = offsets[e] + atomicAdd(&fill[e], 1);
        rowToken[idx] = t;
        assignOf[t * TOPK + j] = idx;
        wArr[idx] = wts[t * TOPK + j];
    }
}

// =====================================================================
// MoE gate+up GEMM (gathered token rows, f16), SiLU*up epilogue -> actH (f16)
// =====================================================================
__global__ void moe_gateup(const _Float16* __restrict__ hn2h, const _Float16* __restrict__ Wg,
                           const _Float16* __restrict__ Wu, const int* __restrict__ rowToken,
                           const int* __restrict__ offsets, const int* __restrict__ counts,
                           _Float16* __restrict__ actH, int e) {
    __shared__ __align__(16) _Float16 sA[16 * SA_LD];
    __shared__ __align__(16) _Float16 sBg[128 * SA_LD];
    __shared__ __align__(16) _Float16 sBu[128 * SA_LD];
    const int cnt = counts[e];
    const int m0 = blockIdx.y * 16;
    if (m0 >= cnt) return;
    const int off = offsets[e];
    const int t = threadIdx.x;
    const int lane = t & 31, wave = t >> 5;
    const int half = lane >> 4, l16 = lane & 15;
    const int nblk = blockIdx.x * 128;
    const int ar = t >> 4, ap = t & 15;
    const int br = t >> 3, bs = t & 7;
    const int arow = m0 + ar;
    const int tokA = rowToken[off + (arow < cnt ? arow : cnt - 1)];
    const _Float16* Bg = Wg + (size_t)e * DM * FF;
    const _Float16* Bu = Wu + (size_t)e * DM * FF;
    const int aOff = l16 * SA_LD + half * 8;
    const int bOff = (wave * 16 + l16) * SA_LD + half * 8;
    v8f accg = {}, accu = {};
    for (int kc = 0; kc < DM; kc += 32) {
        *(unsigned*)(sA + ar * SA_LD + ap * 2) =
            *(const unsigned*)(hn2h + (size_t)tokA * DM + kc + ap * 2);
        stage_b_colmajor(sBg, Bg + (size_t)(kc + br) * FF + nblk + bs * 16, br, bs);
        stage_b_colmajor(sBu, Bu + (size_t)(kc + br) * FF + nblk + bs * 16, br, bs);
        __syncthreads();
        accg = wmma_frag_mma(sA, aOff, sBg, bOff, accg);
        accu = wmma_frag_mma(sA, aOff, sBu, bOff, accu);
        __syncthreads();
    }
    const int col = nblk + wave * 16 + l16;
#pragma unroll
    for (int rr = 0; rr < 8; ++rr) {
        const int row = m0 + rr + half * 8;
        if (row < cnt) {
            const float g = accg[rr], u = accu[rr];
            const float silu = g / (1.f + __expf(-g));
            actH[(size_t)row * FF + col] = (_Float16)(silu * u);
        }
    }
}

// =====================================================================
// MoE down GEMM (dense segment rows, f16), scale by route weight -> yslot (f32)
// =====================================================================
__global__ void moe_down(const _Float16* __restrict__ actH, const _Float16* __restrict__ Wd,
                         const int* __restrict__ offsets, const int* __restrict__ counts,
                         const float* __restrict__ wArr, float* __restrict__ yslot, int e) {
    __shared__ __align__(16) _Float16 sA[16 * SA_LD];
    __shared__ __align__(16) _Float16 sBt[128 * SA_LD];
    const int cnt = counts[e];
    const int m0 = blockIdx.y * 16;
    if (m0 >= cnt) return;
    const int off = offsets[e];
    const int t = threadIdx.x;
    const int lane = t & 31, wave = t >> 5;
    const int half = lane >> 4, l16 = lane & 15;
    const int nblk = blockIdx.x * 128;
    const int ar = t >> 4, ap = t & 15;
    const int br = t >> 3, bs = t & 7;
    const int arow = m0 + ar;
    const size_t aSrc = (size_t)(arow < cnt ? arow : cnt - 1) * FF;
    const _Float16* B = Wd + (size_t)e * FF * DM;
    const int aOff = l16 * SA_LD + half * 8;
    const int bOff = (wave * 16 + l16) * SA_LD + half * 8;
    v8f acc = {};
    for (int kc = 0; kc < FF; kc += 32) {
        *(unsigned*)(sA + ar * SA_LD + ap * 2) =
            *(const unsigned*)(actH + aSrc + kc + ap * 2);
        stage_b_colmajor(sBt, B + (size_t)(kc + br) * DM + nblk + bs * 16, br, bs);
        __syncthreads();
        acc = wmma_frag_mma(sA, aOff, sBt, bOff, acc);
        __syncthreads();
    }
    const int col = nblk + wave * 16 + l16;
#pragma unroll
    for (int rr = 0; rr < 8; ++rr) {
        const int row = m0 + rr + half * 8;
        if (row < cnt) {
            const int assign = off + row;
            yslot[(size_t)assign * DM + col] = acc[rr] * wArr[assign];
        }
    }
}

// ---------------- combine top-k slots (deterministic, no float atomics) ----------------
__global__ void combine_kernel(const float* __restrict__ yslot, const int* __restrict__ assignOf,
                               float* __restrict__ out) {
    const int i = blockIdx.x * blockDim.x + threadIdx.x;
    if (i >= T_TOK * DM) return;
    const int t = i / DM, d = i % DM;
    float s = 0.f;
#pragma unroll
    for (int j = 0; j < TOPK; ++j)
        s += yslot[(size_t)assignOf[t * TOPK + j] * DM + d];
    out[i] = s;
}

// =====================================================================
extern "C" void kernel_launch(void* const* d_in, const int* in_sizes, int n_in,
                              void* d_out, int out_size, void* d_ws, size_t ws_size,
                              hipStream_t stream) {
    const float* hidden   = (const float*)d_in[0];
    const int*   posns    = (const int*)  d_in[1];
    const float* rms1     = (const float*)d_in[2];
    const float* rms2     = (const float*)d_in[3];
    const float* wq       = (const float*)d_in[4];
    const float* wk       = (const float*)d_in[5];
    const float* wv       = (const float*)d_in[6];
    const float* wo       = (const float*)d_in[7];
    const float* router_w = (const float*)d_in[8];
    const float* router_b = (const float*)d_in[9];
    const float* w_gate   = (const float*)d_in[10];
    const float* w_up     = (const float*)d_in[11];
    const float* w_down   = (const float*)d_in[12];
    float* out = (float*)d_out;

    // ---- workspace carve-up ----
    size_t o = 0;
    char* base = (char*)d_ws;
    auto alloc = [&](size_t bytes) -> void* {
        void* p = base + o;
        o += (bytes + 255) & ~(size_t)255;
        return p;
    };
    // f16 weights
    _Float16* wqh = (_Float16*)alloc((size_t)DM * NH * HD * 2);
    _Float16* wkh = (_Float16*)alloc((size_t)DM * NKV * HD * 2);
    _Float16* wvh = (_Float16*)alloc((size_t)DM * NKV * HD * 2);
    _Float16* woh = (_Float16*)alloc((size_t)NH * HD * DM * 2);
    _Float16* wgh = (_Float16*)alloc((size_t)NE * DM * FF * 2);
    _Float16* wuh = (_Float16*)alloc((size_t)NE * DM * FF * 2);
    _Float16* wdh = (_Float16*)alloc((size_t)NE * FF * DM * 2);
    // activations
    _Float16* hn1h = (_Float16*)alloc((size_t)T_TOK * DM * 2);
    float*    qb   = (float*)   alloc((size_t)T_TOK * NH * HD * 4);
    float*    kb   = (float*)   alloc((size_t)T_TOK * NKV * HD * 4);
    float*    vb   = (float*)   alloc((size_t)T_TOK * NKV * HD * 4);
    _Float16* ctxh = (_Float16*)alloc((size_t)T_TOK * NH * HD * 2);
    float*    h2   = (float*)   alloc((size_t)T_TOK * DM * 4);
    float*    hn2  = (float*)   alloc((size_t)T_TOK * DM * 4);
    _Float16* hn2h = (_Float16*)alloc((size_t)T_TOK * DM * 2);
    _Float16* actH = (_Float16*)alloc((size_t)T_TOK * FF * 2);
    float*    yslot= (float*)   alloc((size_t)T_TOK * TOPK * DM * 4);
    // router / gather
    float* logits = (float*)alloc((size_t)T_TOK * NE * 4);
    int*   tkIds  = (int*)  alloc((size_t)T_TOK * TOPK * 4);
    float* tkW    = (float*)alloc((size_t)T_TOK * TOPK * 4);
    int*   counts = (int*)  alloc(64);
    int*   offs   = (int*)  alloc(128);
    int*   fill   = (int*)  alloc(64);
    int*   rowTok = (int*)  alloc((size_t)T_TOK * TOPK * 4);
    int*   asgOf  = (int*)  alloc((size_t)T_TOK * TOPK * 4);
    float* wArr   = (float*)alloc((size_t)T_TOK * TOPK * 4);
    (void)ws_size; (void)n_in; (void)in_sizes; (void)out_size;

    // ---- 0. one-time f16 weight conversion ----
    {
        const int CG = 8192;
        f32_to_f16<<<CG, 256, 0, stream>>>(wq, wqh, (size_t)DM * NH * HD);
        f32_to_f16<<<CG, 256, 0, stream>>>(wk, wkh, (size_t)DM * NKV * HD);
        f32_to_f16<<<CG, 256, 0, stream>>>(wv, wvh, (size_t)DM * NKV * HD);
        f32_to_f16<<<CG, 256, 0, stream>>>(wo, woh, (size_t)NH * HD * DM);
        f32_to_f16<<<CG, 256, 0, stream>>>(w_gate, wgh, (size_t)NE * DM * FF);
        f32_to_f16<<<CG, 256, 0, stream>>>(w_up,   wuh, (size_t)NE * DM * FF);
        f32_to_f16<<<CG, 256, 0, stream>>>(w_down, wdh, (size_t)NE * FF * DM);
    }

    // ---- 1. pre-attention RMSNorm (f16 out for GEMM) ----
    rmsnorm_kernel<<<T_TOK, 256, 0, stream>>>(hidden, rms1, nullptr, hn1h);

    // ---- 2. Q/K/V projections (LDS-staged WMMA) ----
    gemm_h_wmma<false><<<dim3((NH * HD) / 128, T_TOK / 16), 256, 0, stream>>>(
        hn1h, wqh, nullptr, qb, NH * HD, DM);
    gemm_h_wmma<false><<<dim3((NKV * HD) / 128, T_TOK / 16), 256, 0, stream>>>(
        hn1h, wkh, nullptr, kb, NKV * HD, DM);
    gemm_h_wmma<false><<<dim3((NKV * HD) / 128, T_TOK / 16), 256, 0, stream>>>(
        hn1h, wvh, nullptr, vb, NKV * HD, DM);

    // ---- 3. RoPE ----
    {
        int nq = T_TOK * NH * (HD / 2);
        rope_kernel<<<(nq + 255) / 256, 256, 0, stream>>>(qb, NH, posns);
        int nk = T_TOK * NKV * (HD / 2);
        rope_kernel<<<(nk + 255) / 256, 256, 0, stream>>>(kb, NKV, posns);
    }

    // ---- 4. causal GQA attention (f16 ctx out) ----
    attn_kernel<<<dim3(NH, T_TOK), 32, 0, stream>>>(qb, kb, vb, ctxh);

    // ---- 5. output projection + residual (LDS-staged WMMA) ----
    gemm_h_wmma<true><<<dim3(DM / 128, T_TOK / 16), 256, 0, stream>>>(
        ctxh, woh, hidden, h2, DM, NH * HD);

    // ---- 6. post-attention RMSNorm (f32 for router, f16 for MoE GEMM) ----
    rmsnorm_kernel<<<T_TOK, 256, 0, stream>>>(h2, rms2, hn2, hn2h);

    // ---- 7. router -> softmax -> top-k -> expert gather lists ----
    router_kernel<<<(T_TOK * NE + 255) / 256, 256, 0, stream>>>(hn2, router_w, router_b, logits);
    zero_ints<<<1, 64, 0, stream>>>(counts, NE);
    topk_kernel<<<(T_TOK + 255) / 256, 256, 0, stream>>>(logits, tkIds, tkW, counts);
    offsets_kernel<<<1, 1, 0, stream>>>(counts, offs, fill);
    scatter_kernel<<<(T_TOK + 255) / 256, 256, 0, stream>>>(tkIds, tkW, offs, fill,
                                                            rowTok, asgOf, wArr);

    // ---- 8. sparse MoE: only routed (token, expert) pairs computed ----
    for (int e = 0; e < NE; ++e) {
        moe_gateup<<<dim3(FF / 128, T_TOK / 16), 256, 0, stream>>>(
            hn2h, wgh, wuh, rowTok, offs, counts, actH, e);
        moe_down<<<dim3(DM / 128, T_TOK / 16), 256, 0, stream>>>(
            actH, wdh, offs, counts, wArr, yslot, e);
    }

    // ---- 9. deterministic combine of top-k expert outputs ----
    combine_kernel<<<(T_TOK * DM + 255) / 256, 256, 0, stream>>>(yslot, asgOf, out);
}
